// Lin_Transpose_85641647882597
// MI455X (gfx1250) — matrix operand order repaired and checked by reading, not compiled, not run on hardware
//
#include <hip/hip_runtime.h>

// out = x @ M, M = W.T + sum of 9 permuted copies of W  (collapsed algebraically)
// x: 16384x2048 f32, W: 2048x2048 f32, out: 16384x2048 f32
// Strategy: precompute M^T split into bf16 hi/lo (ws, 16 MB), then bf16x3 WMMA GEMM
// (hi*hi + lo*hi + hi*lo with f32 accumulate ~ 2^-16 rel err).

#define K_DIM 2048
#define N_DIM 2048
#define M_ROWS 16384

#define BM 128
#define BN 128
#define BK 32
#define BKP 40   // padded LDS row length in ushorts (80 B, 16B-aligned, skews banks)

typedef __bf16 bf16x16 __attribute__((ext_vector_type(16)));
typedef __bf16 bf16x2  __attribute__((ext_vector_type(2)));
typedef float  f32x8   __attribute__((ext_vector_type(8)));
typedef float  f32x2   __attribute__((ext_vector_type(2)));

union Frag  { uint4 q[2]; bf16x16 v; };
union Pack2 { bf16x2 h; unsigned int u; };

// Split a pair of f32 into bf16 hi (RNE) and bf16 lo (residual), packed as u32.
__device__ __forceinline__ void split_pair(float a, float b,
                                           unsigned int& hi, unsigned int& lo) {
    f32x2 f = {a, b};
    bf16x2 h2 = __builtin_convertvector(f, bf16x2);          // v_cvt_pk_bf16_f32
    f32x2 r = f - __builtin_convertvector(h2, f32x2);
    bf16x2 l2 = __builtin_convertvector(r, bf16x2);
    Pack2 ph, pl;
    ph.h = h2; pl.h = l2;
    hi = ph.u; lo = pl.u;
}

// ---------------------------------------------------------------------------
// Prep: Mt[n][k] = M[k][n] = 2*W[n,k] + 9-permutation sum, split to bf16 hi/lo
// ---------------------------------------------------------------------------
__global__ __launch_bounds__(256) void prep_M(const float* __restrict__ W,
                                              unsigned short* __restrict__ Mt_hi,
                                              unsigned short* __restrict__ Mt_lo) {
    const int N = 2048;
    int idx = blockIdx.x * 256 + threadIdx.x;
    int n = idx >> 11;          // row of Mt — writes coalesced over k
    int k = idx & (N - 1);
    int i = k, j = n;
    int ri = N - 1 - i, rj = N - 1 - j;
    float v = 2.0f * W[j * N + i]              // W.T (reg_out) + W.T term in Ws
            + W[j * N + ri]                    // rot90 k=1
            + W[ri * N + rj]                   // rot90 k=2
            + W[rj * N + i]                    // rot90 k=3
            + W[ri * N + j]                    // W[::-1, :]
            + W[i * N + rj]                    // W[:, ::-1]
            + W[rj * N + ri]                   // W.T[::-1, ::-1]
            + W[((i + N - 1) & (N - 1)) * N + j]   // roll rows by 1
            + W[i * N + ((j + N - 1) & (N - 1))];  // roll cols by 1
    __bf16 h = (__bf16)v;
    float  r = v - (float)h;
    __bf16 l = (__bf16)r;
    Mt_hi[(size_t)n * N + k] = __builtin_bit_cast(unsigned short, h);
    Mt_lo[(size_t)n * N + k] = __builtin_bit_cast(unsigned short, l);
}

// ---------------------------------------------------------------------------
// GEMM: out[16384x2048] = x @ M, bf16x3 via v_wmma_f32_16x16x32_bf16
// Block: 256 threads = 8 waves (2x4), tile 128x128, K step 32, double-buffered LDS
// ---------------------------------------------------------------------------
__global__ __launch_bounds__(256) void gemm_bf16x3(
        const float* __restrict__ x,
        const unsigned short* __restrict__ Bt_hi,   // Mt hi, [N][K] row-major
        const unsigned short* __restrict__ Bt_lo,
        float* __restrict__ out) {
    __shared__ unsigned short sAhi[2][BM * BKP];
    __shared__ unsigned short sAlo[2][BM * BKP];
    __shared__ unsigned short sBhi[2][BN * BKP];
    __shared__ unsigned short sBlo[2][BN * BKP];

    const int tid  = threadIdx.x;
    const int lane = tid & 31;
    const int wid  = tid >> 5;     // 0..7 (wave32)
    const int wm   = wid >> 2;     // 0..1 -> 64 rows each
    const int wn   = wid & 3;      // 0..3 -> 32 cols each
    const int lm   = lane & 15;
    const int lh   = lane >> 4;

    const int row0 = blockIdx.y * BM;
    const int col0 = blockIdx.x * BN;

    // staging thread mapping
    const int ax_c = (tid & 7) * 4;   // float col within BK (float4 per thread)
    const int ax_r = tid >> 3;        // 0..31 ; +32*p covers 128 rows
    const int bx_c = (tid & 3) * 8;   // ushort col within BK (uint4 per thread)
    const int bx_r = tid >> 2;        // 0..63 ; +64*p covers 128 rows

    f32x8 acc[4][2];
    const f32x8 zero = {};
#pragma unroll
    for (int mi = 0; mi < 4; ++mi)
#pragma unroll
        for (int ni = 0; ni < 2; ++ni) acc[mi][ni] = zero;

    struct Stage { float4 xa[4]; uint4 bh[2]; uint4 bl[2]; } st;

    auto fetch = [&](int kt) {
        const int k0 = kt * BK;
#pragma unroll
        for (int p = 0; p < 4; ++p) {
            int r = ax_r + p * 32;
            st.xa[p] = *reinterpret_cast<const float4*>(
                &x[(size_t)(row0 + r) * K_DIM + k0 + ax_c]);
        }
#pragma unroll
        for (int p = 0; p < 2; ++p) {
            int r = bx_r + p * 64;
            st.bh[p] = *reinterpret_cast<const uint4*>(
                &Bt_hi[(size_t)(col0 + r) * K_DIM + k0 + bx_c]);
            st.bl[p] = *reinterpret_cast<const uint4*>(
                &Bt_lo[(size_t)(col0 + r) * K_DIM + k0 + bx_c]);
        }
    };

    auto commit = [&](int b) {
#pragma unroll
        for (int p = 0; p < 4; ++p) {
            int r = ax_r + p * 32;
            const float4 f = st.xa[p];
            uint2 h, l;
            split_pair(f.x, f.y, h.x, l.x);
            split_pair(f.z, f.w, h.y, l.y);
            *reinterpret_cast<uint2*>(&sAhi[b][r * BKP + ax_c]) = h;
            *reinterpret_cast<uint2*>(&sAlo[b][r * BKP + ax_c]) = l;
        }
#pragma unroll
        for (int p = 0; p < 2; ++p) {
            int r = bx_r + p * 64;
            *reinterpret_cast<uint4*>(&sBhi[b][r * BKP + bx_c]) = st.bh[p];
            *reinterpret_cast<uint4*>(&sBlo[b][r * BKP + bx_c]) = st.bl[p];
        }
    };

    auto compute = [&](int b) {
        Frag bh[2], bl[2];
#pragma unroll
        for (int ni = 0; ni < 2; ++ni) {
            int n = wn * 32 + ni * 16 + lm;
            // B frag, lane (n, half): e 0..7 -> K = 8*half+e ; e 8..15 -> K = 16+8*half+(e-8)
            bh[ni].q[0] = *reinterpret_cast<const uint4*>(&sBhi[b][n * BKP + lh * 8]);
            bh[ni].q[1] = *reinterpret_cast<const uint4*>(&sBhi[b][n * BKP + 16 + lh * 8]);
            bl[ni].q[0] = *reinterpret_cast<const uint4*>(&sBlo[b][n * BKP + lh * 8]);
            bl[ni].q[1] = *reinterpret_cast<const uint4*>(&sBlo[b][n * BKP + 16 + lh * 8]);
        }
#pragma unroll
        for (int mi = 0; mi < 4; ++mi) {
            int m = wm * 64 + mi * 16 + lm;
            // A frag, lane (m, half): e 0..7 -> K = 8*half+e ; e 8..15 -> K = 16+8*half+(e-8)
            Frag ah, al;
            ah.q[0] = *reinterpret_cast<const uint4*>(&sAhi[b][m * BKP + lh * 8]);
            ah.q[1] = *reinterpret_cast<const uint4*>(&sAhi[b][m * BKP + 16 + lh * 8]);
            al.q[0] = *reinterpret_cast<const uint4*>(&sAlo[b][m * BKP + lh * 8]);
            al.q[1] = *reinterpret_cast<const uint4*>(&sAlo[b][m * BKP + 16 + lh * 8]);
#pragma unroll
            for (int ni = 0; ni < 2; ++ni) {
                acc[mi][ni] = __builtin_amdgcn_wmma_f32_16x16x32_bf16(
                    false, ah.v, false, bh[ni].v, (short)0, acc[mi][ni], false, false);
                acc[mi][ni] = __builtin_amdgcn_wmma_f32_16x16x32_bf16(
                    false, al.v, false, bh[ni].v, (short)0, acc[mi][ni], false, false);
                acc[mi][ni] = __builtin_amdgcn_wmma_f32_16x16x32_bf16(
                    false, ah.v, false, bl[ni].v, (short)0, acc[mi][ni], false, false);
            }
        }
    };

    const int KT = K_DIM / BK;   // 64

    fetch(0);
    commit(0);
    __syncthreads();

    for (int kt = 0; kt < KT; ++kt) {
        const int cur = kt & 1;
        if (kt + 1 < KT) fetch(kt + 1);     // global loads in flight during WMMAs
        compute(cur);
        if (kt + 1 < KT) commit(cur ^ 1);   // convert + LDS store into other buffer
        __syncthreads();
    }

    // C/D layout: vgpr r, lane (n=lm, half=lh): row = 8*lh + r, col = lm
#pragma unroll
    for (int mi = 0; mi < 4; ++mi) {
#pragma unroll
        for (int ni = 0; ni < 2; ++ni) {
            int col = col0 + wn * 32 + ni * 16 + lm;
#pragma unroll
            for (int r = 0; r < 8; ++r) {
                int row = row0 + wm * 64 + mi * 16 + lh * 8 + r;
                out[(size_t)row * N_DIM + col] = acc[mi][ni][r];
            }
        }
    }
}

extern "C" void kernel_launch(void* const* d_in, const int* in_sizes, int n_in,
                              void* d_out, int out_size, void* d_ws, size_t ws_size,
                              hipStream_t stream) {
    const float* x = (const float*)d_in[0];
    const float* W = (const float*)d_in[1];
    float* out = (float*)d_out;

    unsigned short* Mt_hi = (unsigned short*)d_ws;                    // 8 MB
    unsigned short* Mt_lo = Mt_hi + (size_t)K_DIM * N_DIM;            // 8 MB

    prep_M<<<(K_DIM * N_DIM) / 256, 256, 0, stream>>>(W, Mt_hi, Mt_lo);

    dim3 grid(N_DIM / BN, M_ROWS / BM);   // 16 x 128 blocks
    gemm_bf16x3<<<grid, 256, 0, stream>>>(x, Mt_hi, Mt_lo, out);
}
